// VanillaRNN_33732673143824
// MI455X (gfx1250) — matrix operand order repaired and checked by reading, not compiled, not run on hardware
//
#include <hip/hip_runtime.h>
#include <hip/hip_bf16.h>
#include <math.h>

// WMMA vector types (CDNA5 / gfx1250, wave32)
typedef __attribute__((ext_vector_type(16))) __bf16 v16bf;
typedef __attribute__((ext_vector_type(8)))  float  v8f;
typedef __attribute__((ext_vector_type(4)))  float  v4f;

#define B_TOT 2048
#define T_LEN 128
#define H     512
#define C_OUT 10
#define BM    32          // rows of B handled per block (2 M-tiles)
#define NT    4           // N-tiles (16 cols each) per wave; 8 waves * 4 = 32 tiles = 512 cols
#define HPAD  520         // 512 + 8 bf16 pad -> kills LDS bank conflicts on A loads

union FragU { v16bf v; v4f f[2]; };

__device__ __forceinline__ unsigned short f2bf(float f) {
  unsigned int u = __float_as_uint(f);
  u += 0x7FFFu + ((u >> 16) & 1u);          // round-to-nearest-even
  return (unsigned short)(u >> 16);
}
__device__ __forceinline__ float bf2f(unsigned short b) {
  return __uint_as_float(((unsigned int)b) << 16);
}

// Branch-free tanh on the hardware transcendental pipe:
// tanh(x) = 1 - 2/(exp2(2*log2(e)*x) + 1)
// Saturates exactly: x>>0 -> exp2=inf -> rcp=0 -> 1 ; x<<0 -> exp2=0 -> 1-2 = -1.
// Error way below bf16 storage precision.  No divergent branches (unlike libm tanhf).
__device__ __forceinline__ float fast_tanh(float x) {
  float e = __builtin_amdgcn_exp2f(x * 2.8853900817779268f);  // v_exp_f32
  float r = __builtin_amdgcn_rcpf(e + 1.0f);                  // v_rcp_f32
  return 1.0f - 2.0f * r;
}

// Pre-pass: Whh [H,H] f32 row-major  ->  bf16 column-major image in workspace.
// Column-major => each lane's B-fragment (K {0..7,16..23} / {8..15,24..31} of one N column)
// is two contiguous 16-byte loads, matching the v_wmma B-operand VGPR layout.
__global__ void whh_bf16_colmajor_kernel(const float* __restrict__ Whh,
                                         unsigned short* __restrict__ wsB) {
  int i = blockIdx.x * blockDim.x + threadIdx.x;   // over H*H
  int n = i >> 9;
  int k = i & (H - 1);
  wsB[n * H + k] = f2bf(Whh[k * H + n]);
}

__global__ __launch_bounds__(256)
void rnn_kernel(const float* __restrict__ x,   const float* __restrict__ Win,
                const float* __restrict__ bh,  const float* __restrict__ Wout,
                const float* __restrict__ bout,
                const unsigned short* __restrict__ wsB,
                float* __restrict__ out) {
  __shared__ __align__(16) unsigned short hbuf[2][BM * HPAD];  // double-buffered h (bf16)
  __shared__ float xcol[BM];

  const int tid  = threadIdx.x;
  const int lane = tid & 31;
  const int wave = tid >> 5;              // 8 waves per block
  const int b0   = blockIdx.x * BM;

  // h0 = 0
  for (int i = tid; i < BM * HPAD; i += blockDim.x) hbuf[0][i] = 0;

  const int laneHi  = lane >> 4;          // 0: lanes 0-15, 1: lanes 16-31
  const int col15   = lane & 15;
  const int nbase   = wave * NT;          // 4 N-tiles per wave
  const int rowbase = laneHi * 8;         // C/D layout: VGPR v -> row v + 8*laneHi
  // A layout: lane = row (M), lanes<16 hold K {0..7,16..23}, lanes>=16 {8..15,24..31}
  const int aoff0 = col15 * HPAD + laneHi * 8;          // M-tile 0 (rows 0..15)
  const int aoff1 = (16 + col15) * HPAD + laneHi * 8;   // M-tile 1 (rows 16..31)

  // Per-lane column constants for the 4 N-tiles
  float winr[NT], bhr[NT];
  #pragma unroll
  for (int j = 0; j < NT; ++j) {
    int col = (nbase + j) * 16 + col15;
    winr[j] = Win[col];
    bhr[j]  = bh[col];
  }

  // One 64-bit B base per wave; j/k become compile-time immediate offsets.
  const unsigned short* __restrict__ bwave =
      wsB + (size_t)(nbase * 16 + col15) * H + laneHi * 8;

  int cur = 0;
  for (int t = 0; t < T_LEN - 1; ++t) {
    if (tid < BM) xcol[tid] = x[(b0 + tid) * T_LEN + t];
    __syncthreads();   // xcol ready; h0 zero-init ready on first iter

    float xr[2][8];
    #pragma unroll
    for (int m = 0; m < 2; ++m)
      #pragma unroll
      for (int v = 0; v < 8; ++v) xr[m][v] = xcol[m * 16 + rowbase + v];

    v8f acc[2][NT];
    #pragma unroll
    for (int m = 0; m < 2; ++m)
      #pragma unroll
      for (int j = 0; j < NT; ++j) acc[m][j] = (v8f)(0.0f);

    const unsigned short* __restrict__ hc = hbuf[cur];
    #pragma unroll 2
    for (int k = 0; k < H; k += 32) {
      FragU a0, a1;                       // A fragments for both M-tiles
      a0.f[0] = *(const v4f*)(&hc[aoff0 + k]);
      a0.f[1] = *(const v4f*)(&hc[aoff0 + k + 16]);
      a1.f[0] = *(const v4f*)(&hc[aoff1 + k]);
      a1.f[1] = *(const v4f*)(&hc[aoff1 + k + 16]);
      #pragma unroll
      for (int j = 0; j < NT; ++j) {
        // B fragment: 32x16 bf16 tile of Whh, lane = column, K {0..7,16..23} / {8..15,24..31}.
        const unsigned short* bp = bwave + j * (16 * H) + k;
        FragU bm;
        bm.f[0] = *(const v4f*)(bp);
        bm.f[1] = *(const v4f*)(bp + 16);
        // Reuse each B fragment for both M-tiles: 2 WMMAs per 32B B load.
        acc[0][j] = __builtin_amdgcn_wmma_f32_16x16x32_bf16(
            false, a0.v, false, bm.v, (short)0, acc[0][j], false, false);
        acc[1][j] = __builtin_amdgcn_wmma_f32_16x16x32_bf16(
            false, a1.v, false, bm.v, (short)0, acc[1][j], false, false);
      }
    }

    // h_{t+1} = tanh(acc + x_t*Win + bh) -> other LDS buffer (branch-free tanh)
    unsigned short* __restrict__ hn = hbuf[cur ^ 1];
    #pragma unroll
    for (int m = 0; m < 2; ++m)
      #pragma unroll
      for (int j = 0; j < NT; ++j) {
        int col = (nbase + j) * 16 + col15;
        #pragma unroll
        for (int v = 0; v < 8; ++v) {
          float pre = acc[m][j][v] + xr[m][v] * winr[j] + bhr[j];
          hn[(m * 16 + rowbase + v) * HPAD + col] = f2bf(fast_tanh(pre));
        }
      }
    __syncthreads();   // h_{t+1} complete before it becomes hc
    cur ^= 1;
  }

  // out = h_last @ Wout + bout   (C=10: trivial scalar dot loops)
  const unsigned short* __restrict__ hl = hbuf[cur];
  for (int idx = tid; idx < BM * C_OUT; idx += blockDim.x) {
    int r = idx / C_OUT, c = idx - r * C_OUT;
    float s = bout[c];
    for (int h = 0; h < H; ++h)
      s += bf2f(hl[r * HPAD + h]) * Wout[h * C_OUT + c];
    out[(b0 + r) * C_OUT + c] = s;
  }
}

extern "C" void kernel_launch(void* const* d_in, const int* in_sizes, int n_in,
                              void* d_out, int out_size, void* d_ws, size_t ws_size,
                              hipStream_t stream) {
  const float* x    = (const float*)d_in[0];
  const float* Win  = (const float*)d_in[1];
  const float* Whh  = (const float*)d_in[2];
  const float* bh   = (const float*)d_in[3];
  const float* Wout = (const float*)d_in[4];
  const float* bout = (const float*)d_in[5];
  float* out = (float*)d_out;
  unsigned short* wsB = (unsigned short*)d_ws;   // 512 KB bf16 Whh image

  whh_bf16_colmajor_kernel<<<(H * H) / 256, 256, 0, stream>>>(Whh, wsB);
  rnn_kernel<<<B_TOT / BM, 256, 0, stream>>>(x, Win, bh, Wout, bout, wsB, out);
}
